// AttentionHead_53489522704698
// MI455X (gfx1250) — matrix operand order repaired and checked by reading, not compiled, not run on hardware
//
#include <hip/hip_runtime.h>

#define TDIM 256
#define CDIM 384
#define HDIM 64

typedef __attribute__((ext_vector_type(16))) _Float16 v16h;
typedef __attribute__((ext_vector_type(8)))  _Float16 v8h;
typedef __attribute__((ext_vector_type(8)))  float    v8f;
typedef __attribute__((ext_vector_type(4)))  float    v4f;

union V16H { v16h v; v8h h[2]; };

// Strides / offsets in halves (2-byte units)
#define QH_STR 72          // 64 + 8 pad  (row-major [t][h])
#define VT_STR 264         // 256 + 8 pad (row-major [h][t])
#define PW_STR 264

#define QH_OFF 0
#define KH_OFF 18432       // 256*72
#define VT_OFF 36864       // KH_OFF + 256*72
#define WF_OFF 53760       // VT_OFF + 64*264   (phase-1 overlay)
#define PW_OFF 53760       // phase-2 overlay aliases Wf
#define SMEM_HALVES 127488 // WF_OFF + 3*48*512  -> 254976 bytes (< 320KB WGP LDS)

static __device__ __forceinline__ v8f wmma16(v16h a, v16h b, v8f c) {
  // D = A(16x32 f16) * B(32x16 f16) + C(16x16 f32)
  return __builtin_amdgcn_wmma_f32_16x16x32_f16(false, a, false, b, (short)0, c,
                                                false, false);
}

// A fragment (16x32 f16) from row-major half storage:
// lane = M + 16*hs ; h[m] = row[k0 + hs*16 + m]
static __device__ __forceinline__ v16h ldsA(const _Float16* base, int stride,
                                            int r0, int k0, int lane) {
  const _Float16* p = base + (r0 + (lane & 15)) * stride + k0 + ((lane >> 4) << 4);
  V16H u;
  u.h[0] = *(const v8h*)(p);
  u.h[1] = *(const v8h*)(p + 8);
  return u.v;
}

// B fragment (32x16 f16) from storage where operand column n is a contiguous row:
// lane = n + 16*hs ; h[m] = col_row[k0 + hs*16 + m]
static __device__ __forceinline__ v16h ldsB(const _Float16* base, int stride,
                                            int n0, int k0, int lane) {
  const _Float16* p = base + (n0 + (lane & 15)) * stride + k0 + ((lane >> 4) << 4);
  V16H u;
  u.h[0] = *(const v8h*)(p);
  u.h[1] = *(const v8h*)(p + 8);
  return u.v;
}

// Pre-swizzled W fragment: tile = 32 lanes x 16 contiguous halves
static __device__ __forceinline__ v16h ldWf(const _Float16* Wf, int tb, int lane) {
  const _Float16* p = Wf + tb * 512 + lane * 16;
  V16H u;
  u.h[0] = *(const v8h*)(p);
  u.h[1] = *(const v8h*)(p + 8);
  return u.v;
}

// A fragment of x straight from global f32 (one full 128B line per row per K-block)
static __device__ __forceinline__ v16h gldA(const float* xb, int r0, int k0, int lane) {
  const float* p = xb + (r0 + (lane & 15)) * CDIM + k0 + ((lane >> 4) << 3);
  v4f f0 = *(const v4f*)(p);
  v4f f1 = *(const v4f*)(p + 4);
  v4f f2 = *(const v4f*)(p + 16);
  v4f f3 = *(const v4f*)(p + 20);
  v16h a;
#pragma unroll
  for (int i = 0; i < 4; ++i) {
    a[i]      = (_Float16)f0[i];
    a[4 + i]  = (_Float16)f1[i];
    a[8 + i]  = (_Float16)f2[i];
    a[12 + i] = (_Float16)f3[i];
  }
  return a;
}

extern "C" __global__ __launch_bounds__(256)
void attn_head_fused(const float* __restrict__ x, const float* __restrict__ Wq,
                     const float* __restrict__ Wk, const float* __restrict__ Wv,
                     float* __restrict__ out) {
  __shared__ __align__(16) _Float16 smem[SMEM_HALVES];
  _Float16* qh = smem + QH_OFF;   // [256][72] halves
  _Float16* kh = smem + KH_OFF;   // [256][72]
  _Float16* vT = smem + VT_OFF;   // [64][264]  (v transposed)
  _Float16* Wf = smem + WF_OFF;   // 3 matrices * 48 tiles * 512 halves

  const int tid  = threadIdx.x;
  const int lane = tid & 31;
  const int w    = tid >> 5;
  const int b    = blockIdx.x;
  const int nl   = lane & 15;
  const int hsl  = lane >> 4;

  // ---------- Phase A: pack W into B-fragment layout in LDS ----------
  {
    const float* Wsrc[3] = {Wq, Wk, Wv};
#pragma unroll
    for (int p = 0; p < 3; ++p) {
      const float* W = Wsrc[p];
      _Float16* dst = Wf + p * (48 * 512);
      for (int idx = tid; idx < CDIM * HDIM; idx += 256) {  // coalesced read
        int k  = idx >> 6;          // row in [C]
        int n  = idx & 63;          // col in [H]
        int tb = ((k >> 5) << 2) + (n >> 4);
        int hs = (k >> 3) & 1;
        int m  = (k & 7) | ((k >> 1) & 8);
        int l  = (n & 15) + (hs << 4);
        dst[tb * 512 + l * 16 + m] = (_Float16)W[idx];
      }
    }
  }
  __syncthreads();

  // ---------- Phase B: q/k/v projections (x read exactly once) ----------
  const float* xb = x + (size_t)b * TDIM * CDIM;
#pragma unroll
  for (int rti = 0; rti < 2; ++rti) {
    int rt = 2 * w + rti;
    int r0 = rt * 16;
    v8f acc[3][4] = {};
    for (int kb = 0; kb < 12; ++kb) {
      v16h a = gldA(xb, r0, kb * 32, lane);
#pragma unroll
      for (int p = 0; p < 3; ++p)
#pragma unroll
        for (int n = 0; n < 4; ++n)
          acc[p][n] = wmma16(a, ldWf(Wf + p * (48 * 512), kb * 4 + n, lane), acc[p][n]);
    }
    // scatter C-layout results to LDS as f16
#pragma unroll
    for (int n = 0; n < 4; ++n)
#pragma unroll
      for (int i = 0; i < 8; ++i) {
        int t = r0 + i + 8 * hsl;
        int h = n * 16 + nl;
        qh[t * QH_STR + h] = (_Float16)acc[0][n][i];
        kh[t * QH_STR + h] = (_Float16)acc[1][n][i];
        vT[h * VT_STR + t] = (_Float16)acc[2][n][i];
      }
  }
  __syncthreads();

  // ---------- Phase C: causal attention ----------
  _Float16* Pw = smem + PW_OFF + w * (16 * PW_STR);  // per-wave P staging
#pragma unroll
  for (int pass = 0; pass < 2; ++pass) {
    int rt  = pass ? (15 - w) : w;   // balanced causal work: rt and 15-rt pair
    int r0  = rt * 16;
    int nst = rt + 1;                // s-tiles needed under causal mask

    v16h aq0 = ldsA(qh, QH_STR, r0, 0, lane);
    v16h aq1 = ldsA(qh, QH_STR, r0, 32, lane);

    // scores for the whole row strip, kept in registers
    v8f S[16];
#pragma unroll
    for (int st = 0; st < 16; ++st)
      if (st < nst) {
        v8f c = {};
        c = wmma16(aq0, ldsB(kh, QH_STR, st * 16, 0, lane), c);
        c = wmma16(aq1, ldsB(kh, QH_STR, st * 16, 32, lane), c);
        S[st] = c;
      }

    float mrow[8], lrow[8];
#pragma unroll
    for (int i = 0; i < 8; ++i) { mrow[i] = -1e30f; lrow[i] = 0.f; }

    // scale + causal mask + row max
#pragma unroll
    for (int st = 0; st < 16; ++st)
      if (st < nst) {
#pragma unroll
        for (int i = 0; i < 8; ++i) {
          int t = r0 + i + 8 * hsl;
          int s = st * 16 + nl;
          float sv = S[st][i] * 0.125f;   // H^-0.5
          if (s > t) sv = -1e30f;
          S[st][i] = sv;
          mrow[i]  = fmaxf(mrow[i], sv);
        }
      }
#pragma unroll
    for (int i = 0; i < 8; ++i)
#pragma unroll
      for (int off = 1; off < 16; off <<= 1)
        mrow[i] = fmaxf(mrow[i], __shfl_xor(mrow[i], off, 32));

    // exp + row sum
#pragma unroll
    for (int st = 0; st < 16; ++st)
      if (st < nst) {
#pragma unroll
        for (int i = 0; i < 8; ++i) {
          float e = __expf(S[st][i] - mrow[i]);
          S[st][i] = e;
          lrow[i] += e;
        }
      }
#pragma unroll
    for (int i = 0; i < 8; ++i)
#pragma unroll
      for (int off = 1; off < 16; off <<= 1)
        lrow[i] += __shfl_xor(lrow[i], off, 32);

    // stage P (f16) to LDS in row-major to re-layout C->A; pad to even #tiles
    int nst2 = (nst + 1) & ~1;
#pragma unroll
    for (int st = 0; st < 16; ++st)
      if (st < nst2) {
#pragma unroll
        for (int i = 0; i < 8; ++i) {
          float e = (st < nst) ? S[st][i] : 0.f;
          Pw[(i + 8 * hsl) * PW_STR + st * 16 + nl] = (_Float16)e;
        }
      }

    // out = P @ V
    v8f o[4] = {};
    int nkb = nst2 >> 1;
#pragma unroll
    for (int kb = 0; kb < 8; ++kb)
      if (kb < nkb) {
        v16h aP = ldsA(Pw, PW_STR, 0, kb * 32, lane);
#pragma unroll
        for (int n = 0; n < 4; ++n)
          o[n] = wmma16(aP, ldsB(vT, VT_STR, n * 16, kb * 32, lane), o[n]);
      }

    // normalize and store f32
    float rl[8];
#pragma unroll
    for (int i = 0; i < 8; ++i) rl[i] = 1.0f / lrow[i];

    float* ob = out + (size_t)b * TDIM * HDIM;
#pragma unroll
    for (int n = 0; n < 4; ++n)
#pragma unroll
      for (int i = 0; i < 8; ++i) {
        int t = r0 + i + 8 * hsl;
        int h = n * 16 + nl;
        ob[t * HDIM + h] = o[n][i] * rl[i];
      }
  }
}

extern "C" void kernel_launch(void* const* d_in, const int* in_sizes, int n_in,
                              void* d_out, int out_size, void* d_ws, size_t ws_size,
                              hipStream_t stream) {
  (void)in_sizes; (void)n_in; (void)out_size; (void)d_ws; (void)ws_size;
  const float* x  = (const float*)d_in[0];
  const float* Wq = (const float*)d_in[1];
  const float* Wk = (const float*)d_in[2];
  const float* Wv = (const float*)d_in[3];
  float* out = (float*)d_out;
  hipLaunchKernelGGL(attn_head_fused, dim3(512), dim3(256), 0, stream,
                     x, Wq, Wk, Wv, out);
}
